// PsiU_20126216749470
// MI455X (gfx1250) — matrix operand order repaired and checked by reading, not compiled, not run on hardware
//
#include <hip/hip_runtime.h>
#include <hip/hip_bf16.h>

// ---- problem constants (from reference) ----
#define NXI      512
#define LDIM     128
#define NSTATES  128
#define NINPUTS  64
#define EPSV     1e-3f
#define BATCH    32768
#define HDIM     1152   // 2*NXI + LDIM

typedef __bf16 bf16_t;
typedef __attribute__((ext_vector_type(16))) __bf16 v16bf;
typedef __attribute__((ext_vector_type(8)))  float  v8f;

union FragBF { v16bf v; unsigned u[8]; };

// ---- CDNA5 async global->LDS helpers (ASYNCcnt-tracked) ----
__device__ __forceinline__ unsigned lds_off32(const void* p) {
  return (unsigned)(unsigned long long)p;  // flat->LDS uses addr[31:0]
}
__device__ __forceinline__ void async_ld_b128(unsigned lds_byte_off,
                                              const void* gptr) {
  asm volatile("global_load_async_to_lds_b128 %0, %1, off"
               :: "v"(lds_byte_off), "v"(gptr) : "memory");
}
__device__ __forceinline__ void wait_async0() {
  asm volatile("s_wait_asynccnt 0x0" ::: "memory");
}

// ============================================================
// 1) H = X^T X + eps*I   (fp32, accuracy-critical, tiny cost)
// ============================================================
__global__ __launch_bounds__(256) void hgemm_xtx(const float* __restrict__ X,
                                                 float* __restrict__ H) {
  __shared__ float sA[32][33];
  __shared__ float sB[32][33];
  int tx = threadIdx.x, ty = threadIdx.y;          // 16x16
  int tid = ty * 16 + tx;
  int i0 = blockIdx.x * 32, j0 = blockIdx.y * 32;
  float acc[2][2] = {{0.f, 0.f}, {0.f, 0.f}};
  for (int k0 = 0; k0 < HDIM; k0 += 32) {
    for (int p = tid; p < 32 * 32; p += 256) {
      int r = p >> 5, c = p & 31;
      sA[r][c] = X[(k0 + r) * HDIM + i0 + c];
      sB[r][c] = X[(k0 + r) * HDIM + j0 + c];
    }
    __syncthreads();
#pragma unroll 8
    for (int kk = 0; kk < 32; ++kk) {
      float a0 = sA[kk][2 * ty], a1 = sA[kk][2 * ty + 1];
      float b0 = sB[kk][2 * tx], b1 = sB[kk][2 * tx + 1];
      acc[0][0] += a0 * b0; acc[0][1] += a0 * b1;
      acc[1][0] += a1 * b0; acc[1][1] += a1 * b1;
    }
    __syncthreads();
  }
  for (int ii = 0; ii < 2; ++ii)
    for (int jj = 0; jj < 2; ++jj) {
      int gi = i0 + 2 * ty + ii, gj = j0 + 2 * tx + jj;
      H[gi * HDIM + gj] = acc[ii][jj] + (gi == gj ? EPSV : 0.f);
    }
}

// ============================================================
// 2) Extract constrained REN params; build augmented [E | I].
// ============================================================
__global__ void extract_params(const float* __restrict__ H,
                               const float* __restrict__ Y,
                               float* __restrict__ Fm, float* __restrict__ B1,
                               float* __restrict__ C1, float* __restrict__ D11,
                               float* __restrict__ Lam, float* __restrict__ Aug) {
  int j = blockIdx.x * 16 + threadIdx.x;
  int i = blockIdx.y * 16 + threadIdx.y;
  if (i >= 512 || j >= 512) return;
  Fm[i * 512 + j] = H[(640 + i) * HDIM + j];                       // H31
  float e = 0.5f * (H[i * HDIM + j] + H[(640 + i) * HDIM + 640 + j] +
                    Y[i * 512 + j] - Y[j * 512 + i]);              // E
  Aug[i * 1024 + j] = e;
  Aug[i * 1024 + 512 + j] = (i == j) ? 1.f : 0.f;
  if (j < 128) B1[i * 128 + j] = H[(640 + i) * HDIM + 512 + j];    // H32
  if (i < 128) C1[i * 512 + j] = -H[(512 + i) * HDIM + j];         // -H21
  if (i < 128 && j < 128)
    D11[i * 128 + j] = (j < i) ? -H[(512 + i) * HDIM + 512 + j] : 0.f;
  if (i < 128 && j == 0) Lam[i] = 0.5f * H[(512 + i) * HDIM + 512 + i];
}

// ============================================================
// 3) Gauss-Jordan inversion of E (512x512), single workgroup.
// ============================================================
__global__ __launch_bounds__(512) void gauss_jordan(float* __restrict__ AugIn) {
  volatile float* Aug = AugIn;
  __shared__ float prow[1024];
  int tid = threadIdx.x;  // 512 threads, thread i owns row i
  for (int p = 0; p < 512; ++p) {
    float inv = 1.0f / Aug[p * 1024 + p];
    for (int j = tid; j < 1024; j += 512) prow[j] = Aug[p * 1024 + j] * inv;
    __syncthreads();
    for (int j = tid; j < 1024; j += 512) Aug[p * 1024 + j] = prow[j];
    if (tid != p) {
      float f = Aug[tid * 1024 + p];
      for (int j = 0; j < 1024; ++j) Aug[tid * 1024 + j] -= f * prow[j];
    }
    __threadfence();
    __syncthreads();
  }
}

// ============================================================
// 4) fp32 -> bf16 conversion (strided source rows)
// ============================================================
__global__ void f32_to_bf16(const float* __restrict__ src, int ld, int rows,
                            int cols, bf16_t* __restrict__ dst) {
  int idx = blockIdx.x * blockDim.x + threadIdx.x;
  if (idx >= rows * cols) return;
  int r = idx / cols, c = idx - r * cols;
  dst[idx] = (bf16_t)src[(long)r * ld + c];
}

// ============================================================
// 5) Sequential triangular tanh recurrence, batch-parallel;
//    per-lane eps row lives in a private LDS column (no barriers).
// ============================================================
__global__ __launch_bounds__(128) void tri_scan(const float* __restrict__ pre,
                                                const float* __restrict__ D11,
                                                const float* __restrict__ Lam,
                                                bf16_t* __restrict__ epsB) {
  __shared__ float se[128 * 128];
  int tid = threadIdx.x;
  long b = (long)blockIdx.x * 128 + tid;
  const float* prow = pre + b * 128;
  bf16_t* erow = epsB + b * 128;
  for (int k = 0; k < 128; ++k) {
    float v = prow[k];
    for (int j = 0; j < k; ++j) v += se[j * 128 + tid] * D11[k * 128 + j];
    float e = tanhf(v / Lam[k]);
    se[k * 128 + tid] = e;
    erow[k] = (bf16_t)e;
  }
}

// ============================================================
// 6) Fused 3-source WMMA GEMM with async double-buffered staging:
//    Out[B,N] = sum_s Act_s[B,K_s] @ W_s[N,K_s]^T
//    Workgroup tile 256x64 (8 waves); wave tile 32x64 = 2x4 frags
//    -> 8 v_wmma per K-step per wave, A frag reused 4x, B frag 2x.
//    Tiles staged with global_load_async_to_lds_b128 (ASYNCcnt),
//    LDS pitch 20 uints: 16B-aligned chunks, conflict-free gathers.
// ============================================================
#define APITCH 20
__global__ __launch_bounds__(256) void gemm_bf16_wmma(
    const bf16_t* __restrict__ A0, const bf16_t* __restrict__ W0, int K0,
    const bf16_t* __restrict__ A1, const bf16_t* __restrict__ W1, int K1,
    const bf16_t* __restrict__ A2, const bf16_t* __restrict__ W2, int K2,
    float* __restrict__ outF, bf16_t* __restrict__ outB, int N) {
  __shared__ unsigned sA[2][256 * APITCH];  // 256 rows x 32 bf16 per buffer
  __shared__ unsigned sW[2][64 * APITCH];   // 64 rows x 32 bf16 per buffer
  int tid  = threadIdx.x;
  int wave = tid >> 5, lane = tid & 31;
  int half = lane >> 4, lr = lane & 15;
  long rowTile = (long)blockIdx.x * 256;
  int  colTile = blockIdx.y * 64;

  v8f acc[2][4] = {};

  const bf16_t* As_[3] = {A0, A1, A2};
  const bf16_t* Ws_[3] = {W0, W1, W2};
  int Ks_[3] = {K0, K1, K2};
  int p = 0;  // double-buffer parity, carried across sources

  for (int s = 0; s < 3; ++s) {
    const bf16_t* A = As_[s];
    if (!A) continue;
    const bf16_t* W = Ws_[s];
    int K = Ks_[s];
    int Kh = K >> 1;  // uints per row
    const unsigned* aU = (const unsigned*)A;
    const unsigned* wU = (const unsigned*)W;
    int ktEnd = K >> 5;

    // --- async-stage one 256x32 A tile + 64x32 W tile into buffer pb ---
    auto stage = [&](int kt, int pb) {
      unsigned aBase = lds_off32(&sA[pb][0]);
      unsigned wBase = lds_off32(&sW[pb][0]);
#pragma unroll
      for (int q = 0; q < 4; ++q) {           // A: 1024 16B-chunks / 256 thr
        int c = tid + q * 256;
        int row = c >> 2, cu = (c & 3) << 2;
        const unsigned* g = aU + (rowTile + row) * Kh + kt * 16 + cu;
        async_ld_b128(aBase + (unsigned)(row * APITCH + cu) * 4u, g);
      }
      {                                        // W: 256 16B-chunks / 256 thr
        int row = tid >> 2, cu = (tid & 3) << 2;
        const unsigned* g = wU + (long)(colTile + row) * Kh + kt * 16 + cu;
        async_ld_b128(wBase + (unsigned)(row * APITCH + cu) * 4u, g);
      }
    };

    stage(0, p);
    wait_async0();
    __syncthreads();

    for (int kt = 0; kt < ktEnd; ++kt) {
      if (kt + 1 < ktEnd) stage(kt + 1, p ^ 1);  // prefetch next tile

      // ---- fragment gathers per ISA 7.12.2 VGPR layouts ----
      FragBF a[2], b[4];
#pragma unroll
      for (int fm = 0; fm < 2; ++fm) {
        int m = wave * 32 + fm * 16 + lr;
#pragma unroll
        for (int v = 0; v < 4; ++v)            // K pairs 0..7 (+4 hi half)
          a[fm].u[v] = sA[p][m * APITCH + v + 4 * half];
#pragma unroll
        for (int v = 0; v < 4; ++v)            // K pairs 16..23 (+4 hi half)
          a[fm].u[4 + v] = sA[p][m * APITCH + 8 + v + 4 * half];
      }
#pragma unroll
      for (int fn = 0; fn < 4; ++fn) {
        int n = fn * 16 + lr;
#pragma unroll
        for (int v = 0; v < 8; ++v)            // K pairs 0..7, 16..23 (+4 hi half)
          b[fn].u[v] = sW[p][n * APITCH + v + (v & 4) + 4 * half];
      }
#pragma unroll
      for (int fm = 0; fm < 2; ++fm)
#pragma unroll
        for (int fn = 0; fn < 4; ++fn)
          acc[fm][fn] = __builtin_amdgcn_wmma_f32_16x16x32_bf16(
              false, a[fm].v, false, b[fn].v, (short)0, acc[fm][fn], false,
              false);

      wait_async0();      // next tile fully in LDS
      __syncthreads();    // everyone done reading buffer p
      p ^= 1;
    }
  }

  // ---- epilogue: D layout lane0-15 M=v, lane16-31 M=v+8, N=lane&15 ----
#pragma unroll
  for (int fm = 0; fm < 2; ++fm)
#pragma unroll
    for (int fn = 0; fn < 4; ++fn)
#pragma unroll
      for (int v = 0; v < 8; ++v) {
        long r = rowTile + wave * 32 + fm * 16 + v + 8 * half;
        int cc = colTile + fn * 16 + lr;
        float val = acc[fm][fn][v];
        if (outF) outF[r * N + cc] = val;
        if (outB) outB[r * N + cc] = (bf16_t)val;
      }
}

// ============================================================
// Host-side orchestration
// ============================================================
extern "C" void kernel_launch(void* const* d_in, const int* in_sizes, int n_in,
                              void* d_out, int out_size, void* d_ws,
                              size_t ws_size, hipStream_t stream) {
  (void)in_sizes; (void)n_in; (void)out_size; (void)ws_size;
  const float* w_in  = (const float*)d_in[1];   // (B,1,128)
  const float* xi_in = (const float*)d_in[2];   // (B,1,512)
  const float* X   = (const float*)d_in[3];     // (1152,1152)
  const float* Y   = (const float*)d_in[4];     // (512,512)
  const float* B2  = (const float*)d_in[5];     // (512,128)
  const float* C2  = (const float*)d_in[6];     // (64,512)
  const float* D21 = (const float*)d_in[7];     // (64,128)
  const float* D22 = (const float*)d_in[8];     // (64,128)
  const float* D12 = (const float*)d_in[9];     // (128,128)

  char* ws = (char*)d_ws;
  size_t off = 0;
  auto alloc = [&](size_t bytes) -> void* {
    void* p = ws + off;
    off += (bytes + 255) & ~(size_t)255;
    return p;
  };

  float* H    = (float*)alloc((size_t)HDIM * HDIM * 4);
  float* Fm   = (float*)alloc(512 * 512 * 4);
  float* B1f  = (float*)alloc(512 * 128 * 4);
  float* C1f  = (float*)alloc(128 * 512 * 4);
  float* D11f = (float*)alloc(128 * 128 * 4);
  float* Lamf = (float*)alloc(512 * 4);
  float* Aug  = (float*)alloc(512 * 1024 * 4);
  float* preF = (float*)alloc((size_t)BATCH * 128 * 4);

  bf16_t* xi2b  = (bf16_t*)alloc((size_t)BATCH * 512 * 2);
  bf16_t* w2b   = (bf16_t*)alloc((size_t)BATCH * 128 * 2);
  bf16_t* epsb  = (bf16_t*)alloc((size_t)BATCH * 128 * 2);
  bf16_t* Exib  = (bf16_t*)alloc((size_t)BATCH * 512 * 2);
  bf16_t* Fmb   = (bf16_t*)alloc(512 * 512 * 2);
  bf16_t* B1b   = (bf16_t*)alloc(512 * 128 * 2);
  bf16_t* B2b   = (bf16_t*)alloc(512 * 128 * 2);
  bf16_t* C1b   = (bf16_t*)alloc(128 * 512 * 2);
  bf16_t* D12b  = (bf16_t*)alloc(128 * 128 * 2);
  bf16_t* Einvb = (bf16_t*)alloc(512 * 512 * 2);
  bf16_t* C2b   = (bf16_t*)alloc(64 * 512 * 2);
  bf16_t* D21b  = (bf16_t*)alloc(64 * 128 * 2);
  bf16_t* D22b  = (bf16_t*)alloc(64 * 128 * 2);

  // ---- params: H = X^T X + eps I ; extract ; invert E ----
  hgemm_xtx<<<dim3(HDIM / 32, HDIM / 32), dim3(16, 16), 0, stream>>>(X, H);
  extract_params<<<dim3(32, 32), dim3(16, 16), 0, stream>>>(H, Y, Fm, B1f, C1f,
                                                            D11f, Lamf, Aug);
  gauss_jordan<<<1, 512, 0, stream>>>(Aug);

  // ---- bf16 packing ----
  auto conv = [&](const float* src, int ld, int rows, int cols, bf16_t* dst) {
    int total = rows * cols;
    f32_to_bf16<<<(total + 255) / 256, 256, 0, stream>>>(src, ld, rows, cols,
                                                         dst);
  };
  conv(xi_in, 512, BATCH, 512, xi2b);
  conv(w_in, 128, BATCH, 128, w2b);
  conv(Fm, 512, 512, 512, Fmb);
  conv(B1f, 128, 512, 128, B1b);
  conv(B2, 128, 512, 128, B2b);
  conv(C1f, 512, 128, 512, C1b);
  conv(D12, 128, 128, 128, D12b);
  conv(Aug + 512, 1024, 512, 512, Einvb);  // right half of [E|I] -> E^-1
  conv(C2, 512, 64, 512, C2b);
  conv(D21, 128, 64, 128, D21b);
  conv(D22, 128, 64, 128, D22b);

  dim3 blk(256);
  // pre = xi2 @ C1^T + w2 @ D12^T   (B x 128)
  gemm_bf16_wmma<<<dim3(BATCH / 256, 128 / 64), blk, 0, stream>>>(
      xi2b, C1b, 512, w2b, D12b, 128, nullptr, nullptr, 0, preF, nullptr, 128);

  // triangular tanh recurrence -> eps (bf16)
  tri_scan<<<BATCH / 128, 128, 0, stream>>>(preF, D11f, Lamf, epsb);

  // E_xi = xi2 @ Fm^T + eps @ B1^T + w2 @ B2^T  (B x 512, bf16 out)
  gemm_bf16_wmma<<<dim3(BATCH / 256, 512 / 64), blk, 0, stream>>>(
      xi2b, Fmb, 512, epsb, B1b, 128, w2b, B2b, 128, nullptr, Exib, 512);

  float* uOut  = (float*)d_out;                       // (B,1,64) first
  float* xiOut = (float*)d_out + (size_t)BATCH * 64;  // then (B,1,512)

  // u = xi2 @ C2^T + eps @ D21^T + w2 @ D22^T  (B x 64)
  gemm_bf16_wmma<<<dim3(BATCH / 256, 64 / 64), blk, 0, stream>>>(
      xi2b, C2b, 512, epsb, D21b, 128, w2b, D22b, 128, uOut, nullptr, 64);

  // xi_ = E_xi @ inv(E)^T  (B x 512)
  gemm_bf16_wmma<<<dim3(BATCH / 256, 512 / 64), blk, 0, stream>>>(
      Exib, Einvb, 512, nullptr, nullptr, 0, nullptr, nullptr, 0, xiOut,
      nullptr, 512);
}
